// CPCSpatialLoss_29497835389667
// MI455X (gfx1250) — compile-verified, matrix operand order checked by reading, not run on hardware
//
#include <hip/hip_runtime.h>

typedef __attribute__((ext_vector_type(16))) __bf16 v16bf;
typedef __attribute__((ext_vector_type(4)))  __bf16 v4bf;
typedef __attribute__((ext_vector_type(8)))  float  v8f;
typedef __attribute__((ext_vector_type(4)))  float  v4f;

#define NPTS   256
#define DDIM   (64 * 64 * 64)       // 262144
#define TILE   128                  // macro tile (M and N)
#define SPLITK 64
#define KC     (DDIM / SPLITK)      // 4096 per split
#define KSTEP  32

union FragU {
    uint4 u[2];
    v16bf v;
};

// f32x4 -> packed bf16x4 (RNE) via hardware v_cvt_pk_bf16_f32.
__device__ __forceinline__ uint2 pack4_bf16(float4 v) {
    v4f f = {v.x, v.y, v.z, v.w};
    v4bf b = __builtin_convertvector(f, v4bf);
    return __builtin_bit_cast(uint2, b);
}

// Grid: (2 nTiles, 2 mTiles, 64 splits). Block: 256 threads = 8 waves (wave32),
// arranged 4(M) x 2(N); each wave owns a 32x64 patch = eight 16x16 accumulators.
// Double-buffered LDS staging with the barrier at the END of each step so the
// next-stage global_load_b128s overlap the ds_load/WMMA block.
__global__ __launch_bounds__(256)
void gemm_bf16_wmma(const float* __restrict__ V1, const float* __restrict__ V2,
                    float* __restrict__ partial) {
    __shared__ __align__(16) unsigned short lA[2][TILE][KSTEP];  // 2 x 8 KB
    __shared__ __align__(16) unsigned short lB[2][TILE][KSTEP];  // 2 x 8 KB

    const int tn    = blockIdx.x;
    const int tm    = blockIdx.y;
    const int split = blockIdx.z;
    const int tid   = threadIdx.x;
    const int lane  = tid & 31;
    const int wave  = tid >> 5;
    const int wm    = wave >> 1;        // 0..3: M strip (32 rows)
    const int wn    = wave & 1;         // 0..1: N strip (64 cols)

    const long  kBase = (long)split * KC;
    const float* __restrict__ aBase = V1 + (long)(tm * TILE) * DDIM + kBase;
    const float* __restrict__ bBase = V2 + (long)(tn * TILE) * DDIM + kBase;

    v8f acc[2][4];
#pragma unroll
    for (int ms = 0; ms < 2; ++ms)
#pragma unroll
        for (int ns = 0; ns < 4; ++ns)
#pragma unroll
            for (int r = 0; r < 8; ++r) acc[ms][ns][r] = 0.0f;

    // 16-bit A/B fragment layout (wave32): lanes 0-15 -> row/col 0-15, K 0..7 & 16..23;
    // lanes 16-31 -> same rows, K 8..15 & 24..31.
    const int frow = lane & 15;
    const int fkb  = (lane >> 4) * 8;

    // Cooperative staging: 128x32 fp32 per matrix per step = 1024 float4;
    // 256 threads x 4 float4 each.
    int srow[4], sc4[4];
#pragma unroll
    for (int q = 0; q < 4; ++q) {
        int f4 = q * 256 + tid;          // 0..1023
        srow[q] = f4 >> 3;               // 0..127
        sc4[q]  = f4 & 7;                // float4 slot in the 32-wide row
    }

    float4 ra[4], rb[4];
    uint2  ca[4], cb[4];

    auto stage_load = [&](int k0) {
#pragma unroll
        for (int q = 0; q < 4; ++q) {
            ra[q] = *(const float4*)(aBase + (long)srow[q] * DDIM + k0 + sc4[q] * 4);
            rb[q] = *(const float4*)(bBase + (long)srow[q] * DDIM + k0 + sc4[q] * 4);
        }
    };
    auto stage_convert = [&]() {
#pragma unroll
        for (int q = 0; q < 4; ++q) {
            ca[q] = pack4_bf16(ra[q]);
            cb[q] = pack4_bf16(rb[q]);
        }
    };
    auto stage_store = [&](int buf) {
#pragma unroll
        for (int q = 0; q < 4; ++q) {
            *(uint2*)&lA[buf][srow[q]][sc4[q] * 4] = ca[q];
            *(uint2*)&lB[buf][srow[q]][sc4[q] * 4] = cb[q];
        }
    };

    // Prologue: stage buffer 0 and make it visible.
    stage_load(0);
    stage_convert();
    stage_store(0);
    __syncthreads();
    int buf = 0;

    for (int k0 = 0; k0 < KC; k0 += KSTEP) {
        const bool has_next = (k0 + KSTEP) < KC;
        if (has_next) {
            stage_load(k0 + KSTEP);  // in flight across the whole WMMA block
            __builtin_prefetch(aBase + (long)(tid >> 1) * DDIM + k0 + 2 * KSTEP + (tid & 1) * 16, 0, 1);
            __builtin_prefetch(bBase + (long)(tid >> 1) * DDIM + k0 + 2 * KSTEP + (tid & 1) * 16, 0, 1);
        }

        // Consume current buffer: 12 ds_load_b128 -> 8 v_wmma.
        FragU fa[2];
#pragma unroll
        for (int ms = 0; ms < 2; ++ms) {
            const unsigned short* rp = &lA[buf][wm * 32 + ms * 16 + frow][0];
            fa[ms].u[0] = *(const uint4*)(rp + fkb);
            fa[ms].u[1] = *(const uint4*)(rp + fkb + 16);
        }
#pragma unroll
        for (int ns = 0; ns < 4; ++ns) {
            FragU fb;
            const unsigned short* rp = &lB[buf][wn * 64 + ns * 16 + frow][0];
            fb.u[0] = *(const uint4*)(rp + fkb);
            fb.u[1] = *(const uint4*)(rp + fkb + 16);
#pragma unroll
            for (int ms = 0; ms < 2; ++ms)
                acc[ms][ns] = __builtin_amdgcn_wmma_f32_16x16x32_bf16(
                    false, fa[ms].v, false, fb.v, (short)0, acc[ms][ns], false, false);
        }

        // First use of the loaded registers is HERE, after the WMMAs, so the
        // s_wait_loadcnt lands after the compute, not before the barrier.
        if (has_next) {
            stage_convert();
            stage_store(buf ^ 1);
        }
        __syncthreads();  // publish buf^1; next iteration consumes it
        buf ^= 1;
    }

    // Write partials: partial[split][m][n]; exclusive region per block/wave.
    // C/D layout: VGPR r -> M = r (lanes 0-15) or r+8 (lanes 16-31); N = lane&15.
    const int mHalf = (lane >> 4) * 8;
    const int nCol  = lane & 15;
    float* outBase = partial + ((long)split * NPTS + (tm * TILE + wm * 32)) * NPTS
                   + tn * TILE + wn * 64;
#pragma unroll
    for (int ms = 0; ms < 2; ++ms)
#pragma unroll
        for (int ns = 0; ns < 4; ++ns)
#pragma unroll
            for (int r = 0; r < 8; ++r)
                outBase[(long)(ms * 16 + mHalf + r) * NPTS + ns * 16 + nCol] = acc[ms][ns][r];
}

// 64 blocks x 256 threads: fold the 64 split-K partials, scale by 1/D, apply
// the numerically-stable softplus form of the clipped BCE, block-reduce.
__global__ __launch_bounds__(256)
void loss_partial(const float* __restrict__ partial, float* __restrict__ bsum) {
    const int tid  = threadIdx.x;
    const int idx0 = blockIdx.x * 1024;
    float s = 0.0f;
#pragma unroll
    for (int t = 0; t < 1024; t += 256) {
        int ij = idx0 + t + tid;
        int i = ij >> 8, j = ij & 255;
        float z = 0.0f;
        for (int sp = 0; sp < SPLITK; ++sp)
            z += partial[(long)sp * (NPTS * NPTS) + ij];
        float x = z * (1.0f / (float)DDIM);
        // -log(sigmoid(x)) = softplus(-x); -log(1-sigmoid(x)) = softplus(x)
        float t0 = (i == j) ? -x : x;
        s += fmaxf(t0, 0.0f) + log1pf(__expf(-fabsf(t0)));
    }
    __shared__ float red[256];
    red[tid] = s;
    __syncthreads();
#pragma unroll
    for (int off = 128; off > 0; off >>= 1) {
        if (tid < off) red[tid] += red[tid + off];
        __syncthreads();
    }
    if (tid == 0) bsum[blockIdx.x] = red[0];
}

__global__ __launch_bounds__(64)
void loss_final(const float* __restrict__ bsum, float* __restrict__ out) {
    __shared__ float red[64];
    const int tid = threadIdx.x;
    red[tid] = bsum[tid];
    __syncthreads();
#pragma unroll
    for (int off = 32; off > 0; off >>= 1) {
        if (tid < off) red[tid] += red[tid + off];
        __syncthreads();
    }
    if (tid == 0) out[0] = red[0] * (1.0f / (float)(NPTS * NPTS));
}

extern "C" void kernel_launch(void* const* d_in, const int* in_sizes, int n_in,
                              void* d_out, int out_size, void* d_ws, size_t ws_size,
                              hipStream_t stream) {
    (void)in_sizes; (void)n_in; (void)out_size; (void)ws_size;
    const float* V1 = (const float*)d_in[0];
    const float* V2 = (const float*)d_in[1];
    float* out = (float*)d_out;

    float* partial = (float*)d_ws;                         // 64 * 256 * 256 f32 = 16 MB
    float* bsum    = partial + (long)SPLITK * NPTS * NPTS; // + 64 f32

    dim3 ggrid(NPTS / TILE, NPTS / TILE, SPLITK);          // (2,2,64)
    gemm_bf16_wmma<<<ggrid, 256, 0, stream>>>(V1, V2, partial);
    loss_partial<<<64, 256, 0, stream>>>(partial, bsum);
    loss_final<<<1, 64, 0, stream>>>(bsum, out);
}
